// BertSelfAttention_43284680409744
// MI455X (gfx1250) — matrix operand order repaired and checked by reading, not compiled, not run on hardware
//
#include <hip/hip_runtime.h>

typedef __attribute__((ext_vector_type(16))) __bf16 bf16x16;
typedef __attribute__((ext_vector_type(8)))  __bf16 bf16x8;
typedef __attribute__((ext_vector_type(8)))  float  f32x8;
typedef __attribute__((ext_vector_type(4)))  float  f32x4;

#define NHEADS 16
#define HDIM   64
#define SEQ    2048
#define BATCH  4
#define HID    1024
#define MTOT   (BATCH * SEQ)   // 8192

#define LOG2E      1.44269504088896340736f
#define SCALE_LOG2 (0.125f * LOG2E)   // 1/sqrt(64) folded into exp2 domain

__device__ __forceinline__ f32x8 wmma_bf16(bf16x16 a, bf16x16 b, f32x8 c) {
  // D = A(16x32 bf16) * B(32x16 bf16) + C(16x16 f32)
  return __builtin_amdgcn_wmma_f32_16x16x32_bf16(false, a, false, b, (short)0, c,
                                                 false, false);
}

// gfx1250 async copy: global -> LDS, 16 bytes per lane (tracked by ASYNCcnt)
__device__ __forceinline__ void async_copy_b128(unsigned lds_off,
                                                const __bf16* gptr) {
  asm volatile("global_load_async_to_lds_b128 %0, %1, off"
               :
               : "v"(lds_off), "v"(gptr)
               : "memory");
}
__device__ __forceinline__ void wait_async0() {
  asm volatile("s_wait_asynccnt 0" ::: "memory");
}

// ---------------------------------------------------------------------------
// Kernel 0: elementwise f32 -> bf16 (8 elements / thread, b128 in, b128 out)
// ---------------------------------------------------------------------------
__global__ __launch_bounds__(256) void cvt_bf16_kernel(
    const float* __restrict__ src, __bf16* __restrict__ dst, int n) {
  const int i = (blockIdx.x * 256 + threadIdx.x) * 8;
  if (i + 8 <= n) {
    f32x4 a = *(const f32x4*)(src + i);
    f32x4 b = *(const f32x4*)(src + i + 4);
    bf16x8 o;
#pragma unroll
    for (int e = 0; e < 4; ++e) {
      o[e]     = (__bf16)a[e];
      o[4 + e] = (__bf16)b[e];
    }
    *(bf16x8*)(dst + i) = o;
  }
}

// ---------------------------------------------------------------------------
// Kernel 1 helpers (BM=256, BN=64, BK=32; wave owns 32 rows x 64 cols)
// ---------------------------------------------------------------------------
__device__ __forceinline__ void stage_tiles(unsigned xl, unsigned wl,
                                            const __bf16* xg,
                                            const __bf16* wg) {
  // X: one full 32-element row per thread (4 x b128); W: 8 elems (1 x b128)
  async_copy_b128(xl +  0, xg +  0);
  async_copy_b128(xl + 16, xg +  8);
  async_copy_b128(xl + 32, xg + 16);
  async_copy_b128(xl + 48, xg + 24);
  async_copy_b128(wl, wg);
}

__device__ __forceinline__ void gemm_tile(const __bf16 Xb[][32],
                                          const __bf16 Wb[][32], int wave,
                                          int r, int h, f32x8 acc[2][4]) {
  // A fragments for the wave's two 16-row strips (BK==32 == one WMMA K)
  bf16x16 A[2];
#pragma unroll
  for (int ms = 0; ms < 2; ++ms) {
    bf16x8 a0 = *(const bf16x8*)&Xb[wave * 32 + ms * 16 + r][h * 16];
    bf16x8 a1 = *(const bf16x8*)&Xb[wave * 32 + ms * 16 + r][h * 16 + 8];
#pragma unroll
    for (int e = 0; e < 8; ++e) { A[ms][e] = a0[e]; A[ms][8 + e] = a1[e]; }
  }
  // preload all B fragments; each feeds TWO WMMAs (1.5 ds loads / WMMA)
  bf16x16 Bf[4];
#pragma unroll
  for (int nt = 0; nt < 4; ++nt)
    Bf[nt] = *(const bf16x16*)&Wb[nt * 16 + r][h * 16];
#pragma unroll
  for (int nt = 0; nt < 4; ++nt)
#pragma unroll
    for (int ms = 0; ms < 2; ++ms)
      acc[ms][nt] = wmma_bf16(A[ms], Bf[nt], acc[ms][nt]);
}

// ---------------------------------------------------------------------------
// Kernel 1: fused QKV projection on bf16 inputs, double-buffered async LDS.
// out[m,n] = X[m,:] . W[n,:] + bias[n];   grid = (M/256, N/64, 3), 256 thr.
// z=0 -> Q bf16 [bh][s][d], z=1 -> K bf16 [bh][s][d], z=2 -> V^T bf16 [bh][d][s]
// ---------------------------------------------------------------------------
__global__ __launch_bounds__(256) void qkv_gemm_kernel(
    const __bf16* __restrict__ Xbf, const __bf16* __restrict__ Wbf,
    const float* __restrict__ bq, const float* __restrict__ bk,
    const float* __restrict__ bv,
    __bf16* __restrict__ Qbf, __bf16* __restrict__ Kbf, __bf16* __restrict__ Vt)
{
  __shared__ __align__(16) __bf16 Xs[2][256][32];  // 32 KB (double buffer)
  __shared__ __align__(16) __bf16 Ws2[2][64][32];  //  8 KB (double buffer)

  const int z = blockIdx.z;
  const __bf16* __restrict__ W = Wbf + (size_t)z * HID * HID;
  const float* __restrict__ bias = (z == 0) ? bq : ((z == 1) ? bk : bv);

  const int m0   = blockIdx.x * 256;
  const int n0   = blockIdx.y * 64;
  const int tid  = threadIdx.x;
  const int lane = tid & 31;
  const int wave = tid >> 5;
  const int r    = lane & 15;
  const int h    = lane >> 4;

  f32x8 acc[2][4];
#pragma unroll
  for (int ms = 0; ms < 2; ++ms)
#pragma unroll
    for (int nt = 0; nt < 4; ++nt)
#pragma unroll
      for (int e = 0; e < 8; ++e) acc[ms][nt][e] = 0.0f;

  // per-thread staging slots: X row = tid; W row = tid>>2, 8-col segment
  const int wrow = tid >> 2, wcs = (tid & 3) * 8;
  const __bf16* xg = Xbf + (size_t)(m0 + tid) * HID;
  const __bf16* wg = W   + (size_t)(n0 + wrow) * HID + wcs;
  const unsigned xl0 = (unsigned)(size_t)&Xs[0][tid][0];
  const unsigned wl0 = (unsigned)(size_t)&Ws2[0][wrow][wcs];
  const unsigned XBUF = 256 * 32 * 2;  // bytes per X buffer
  const unsigned WBUF = 64 * 32 * 2;   // bytes per W buffer

  // prologue: stage first tile into buffer 0
  stage_tiles(xl0, wl0, xg, wg);

  for (int k0 = 0; k0 < HID; k0 += 64) {
    // ---- buffer 0: wait copies, overlap next stage with compute ----
    wait_async0();
    __syncthreads();
    stage_tiles(xl0 + XBUF, wl0 + WBUF, xg + k0 + 32, wg + k0 + 32);
    gemm_tile(Xs[0], Ws2[0], wave, r, h, acc);

    // ---- buffer 1 ----
    wait_async0();
    __syncthreads();
    if (k0 + 64 < HID)
      stage_tiles(xl0, wl0, xg + k0 + 64, wg + k0 + 64);
    gemm_tile(Xs[1], Ws2[1], wave, r, h, acc);
  }

  // ---- bias + store (D layout: VGPR j -> row j + 8*h, col = r) ----
#pragma unroll
  for (int ms = 0; ms < 2; ++ms)
#pragma unroll
    for (int nt = 0; nt < 4; ++nt) {
      const int n_g  = n0 + nt * 16 + r;
      const float bb = bias[n_g];
      const int head = n_g >> 6;
      const int d    = n_g & 63;
      const int mbase = m0 + wave * 32 + ms * 16 + 8 * h;
      const int bi    = m0 >> 11;          // 256-row block never crosses batch
      const int bh    = bi * NHEADS + head;
      if (z == 2) {
        // V^T: lane's 8 rows are consecutive s -> contiguous in [d][s]
        bf16x8 pk;
#pragma unroll
        for (int j = 0; j < 8; ++j) pk[j] = (__bf16)(acc[ms][nt][j] + bb);
        const int s0 = mbase & 2047;
        *(bf16x8*)&Vt[((size_t)bh * HDIM + d) * SEQ + s0] = pk;
      } else {
        __bf16* dst = (z == 0) ? Qbf : Kbf;
#pragma unroll
        for (int j = 0; j < 8; ++j) {
          const int s = (mbase + j) & 2047;
          dst[((size_t)bh * SEQ + s) * HDIM + d] = (__bf16)(acc[ms][nt][j] + bb);
        }
      }
    }
}

// ---------------------------------------------------------------------------
// Kernel 2: flash attention.  grid = (SEQ/128, BATCH*NHEADS), block = 256.
// Each wave owns a 16-row Q strip; loops over S in 64-wide chunks.
// Softmax runs in the exp2 domain (v_exp_f32 native).
// ---------------------------------------------------------------------------
__global__ __launch_bounds__(256) void attn_kernel(
    const __bf16* __restrict__ Qbf, const __bf16* __restrict__ Kbf,
    const __bf16* __restrict__ Vt, const float* __restrict__ mask,
    float* __restrict__ out)
{
  __shared__ __align__(32) __bf16 Plds[8][16][64];  // 16 KB, wave-private slabs

  const int tid  = threadIdx.x;
  const int lane = tid & 31;
  const int wave = tid >> 5;
  const int r    = lane & 15;
  const int h    = lane >> 4;
  const int bh   = blockIdx.y;
  const int b    = bh >> 4;
  const int head = bh & 15;
  const int q0   = blockIdx.x * 128 + wave * 16;

  // ---- Q fragments resident in registers (full head_dim = 2 K-chunks) ----
  bf16x16 qa[2];
  {
    const __bf16* qp = Qbf + ((size_t)bh * SEQ + q0 + r) * HDIM;
#pragma unroll
    for (int dc = 0; dc < 2; ++dc) {
      bf16x8 c0 = *(const bf16x8*)(qp + dc * 32 + h * 16);
      bf16x8 c1 = *(const bf16x8*)(qp + dc * 32 + h * 16 + 8);
#pragma unroll
      for (int e = 0; e < 8; ++e) {
        qa[dc][e]     = c0[e];
        qa[dc][8 + e] = c1[e];
      }
    }
  }

  f32x8 o[4];
  float mrow[8], lrow[8];
#pragma unroll
  for (int nt = 0; nt < 4; ++nt)
#pragma unroll
    for (int e = 0; e < 8; ++e) o[nt][e] = 0.0f;
#pragma unroll
  for (int j = 0; j < 8; ++j) { mrow[j] = -1e30f; lrow[j] = 0.0f; }

  const float* __restrict__ mp = mask + (size_t)b * SEQ;
  const __bf16* __restrict__ kbase = Kbf + (size_t)bh * SEQ * HDIM;
  const __bf16* __restrict__ vbase = Vt + (size_t)bh * HDIM * SEQ;

  for (int sk = 0; sk < SEQ; sk += 64) {
    // ---- scores: four 16-wide tiles, 2 WMMAs each over head_dim ----
    f32x8 st[4];
#pragma unroll
    for (int t = 0; t < 4; ++t) {
      const __bf16* kp = kbase + (size_t)(sk + t * 16 + r) * HDIM;
      f32x8 s;
#pragma unroll
      for (int e = 0; e < 8; ++e) s[e] = 0.0f;
#pragma unroll
      for (int dc = 0; dc < 2; ++dc) {
        bf16x16 kb = *(const bf16x16*)(kp + dc * 32 + h * 16);
        s = wmma_bf16(qa[dc], kb, s);
      }
      const float mval = mp[sk + t * 16 + r] * LOG2E;  // additive mask
#pragma unroll
      for (int e = 0; e < 8; ++e) s[e] = s[e] * SCALE_LOG2 + mval;
      st[t] = s;
    }

    // ---- online softmax (exp2 domain); rows live in 16-lane halves ----
#pragma unroll
    for (int j = 0; j < 8; ++j) {
      float x = fmaxf(fmaxf(st[0][j], st[1][j]), fmaxf(st[2][j], st[3][j]));
#pragma unroll
      for (int off = 8; off >= 1; off >>= 1)
        x = fmaxf(x, __shfl_xor(x, off, 32));
      const float mnew  = fmaxf(mrow[j], x);
      const float alpha = exp2f(mrow[j] - mnew);
      mrow[j] = mnew;
      float rs = 0.0f;
#pragma unroll
      for (int t = 0; t < 4; ++t) {
        const float p = exp2f(st[t][j] - mnew);
        st[t][j] = p;
        rs += p;
      }
#pragma unroll
      for (int off = 8; off >= 1; off >>= 1)
        rs += __shfl_xor(rs, off, 32);
      lrow[j] = lrow[j] * alpha + rs;
      o[0][j] *= alpha; o[1][j] *= alpha; o[2][j] *= alpha; o[3][j] *= alpha;
    }

    // ---- P: D-layout -> A-layout via wave-private LDS slab ----
#pragma unroll
    for (int t = 0; t < 4; ++t)
#pragma unroll
      for (int j = 0; j < 8; ++j)
        Plds[wave][j + 8 * h][r + t * 16] = (__bf16)st[t][j];

    bf16x16 pf[2];
#pragma unroll
    for (int cc = 0; cc < 2; ++cc) {
      bf16x8 c0 = *(const bf16x8*)&Plds[wave][r][cc * 32 + h * 16];
      bf16x8 c1 = *(const bf16x8*)&Plds[wave][r][cc * 32 + h * 16 + 8];
#pragma unroll
      for (int e = 0; e < 8; ++e) { pf[cc][e] = c0[e]; pf[cc][8 + e] = c1[e]; }
    }

    // ---- ctx += P @ V  (V^T rows are K-contiguous B fragments) ----
#pragma unroll
    for (int cc = 0; cc < 2; ++cc)
#pragma unroll
      for (int nt = 0; nt < 4; ++nt) {
        bf16x16 vb = *(const bf16x16*)(vbase + (size_t)(nt * 16 + r) * SEQ +
                                       sk + cc * 32 + h * 16);
        o[nt] = wmma_bf16(pf[cc], vb, o[nt]);
      }
  }

  // ---- normalize and store f32 [b, s, head*64 + d] ----
  float inv[8];
#pragma unroll
  for (int j = 0; j < 8; ++j) inv[j] = 1.0f / lrow[j];
#pragma unroll
  for (int nt = 0; nt < 4; ++nt)
#pragma unroll
    for (int j = 0; j < 8; ++j) {
      const int s = q0 + j + 8 * h;
      out[((size_t)b * SEQ + s) * HID + head * HDIM + nt * 16 + r] =
          o[nt][j] * inv[j];
    }
}

// ---------------------------------------------------------------------------
extern "C" void kernel_launch(void* const* d_in, const int* in_sizes, int n_in,
                              void* d_out, int out_size, void* d_ws,
                              size_t ws_size, hipStream_t stream) {
  (void)in_sizes; (void)n_in; (void)out_size; (void)ws_size;
  const float* hs   = (const float*)d_in[0];
  const float* mask = (const float*)d_in[1];
  const float* Wq   = (const float*)d_in[2];
  const float* bq   = (const float*)d_in[3];
  const float* Wk   = (const float*)d_in[4];
  const float* bk   = (const float*)d_in[5];
  const float* Wv   = (const float*)d_in[6];
  const float* bv   = (const float*)d_in[7];
  float* out = (float*)d_out;

  const size_t elems = (size_t)BATCH * NHEADS * SEQ * HDIM;  // 8.4M (== M*HID)
  const size_t welems = (size_t)HID * HID;                   // 1M
  __bf16* Qbf = (__bf16*)d_ws;
  __bf16* Kbf = Qbf + elems;
  __bf16* Vt  = Kbf + elems;
  __bf16* Xbf = Vt + elems;
  __bf16* Wbf = Xbf + elems;  // 3 weight matrices, concatenated

  // ---- stage 0: convert inputs to bf16 once ----
  cvt_bf16_kernel<<<(int)(elems / 2048), 256, 0, stream>>>(hs, Xbf, (int)elems);
  cvt_bf16_kernel<<<(int)(welems / 2048), 256, 0, stream>>>(Wq, Wbf + 0 * welems,
                                                            (int)welems);
  cvt_bf16_kernel<<<(int)(welems / 2048), 256, 0, stream>>>(Wk, Wbf + 1 * welems,
                                                            (int)welems);
  cvt_bf16_kernel<<<(int)(welems / 2048), 256, 0, stream>>>(Wv, Wbf + 2 * welems,
                                                            (int)welems);

  // ---- stage 1: fused QKV projection ----
  dim3 g1(MTOT / 256, HID / 64, 3);  // 32 x 16 x 3
  qkv_gemm_kernel<<<g1, 256, 0, stream>>>(Xbf, Wbf, bq, bk, bv, Qbf, Kbf, Vt);

  // ---- stage 2: flash attention ----
  dim3 g2(SEQ / 128, BATCH * NHEADS);  // 16 x 64
  attn_kernel<<<g2, 256, 0, stream>>>(Qbf, Kbf, Vt, mask, out);
}
